// ConvNet_29807073034785
// MI455X (gfx1250) — matrix operand order repaired and checked by reading, not compiled, not run on hardware
//
#include <hip/hip_runtime.h>
#include <hip/hip_bf16.h>

typedef _Float16 half_t;
typedef __attribute__((ext_vector_type(8)))  _Float16 v8h;
typedef __attribute__((ext_vector_type(16))) _Float16 v16h;
typedef __attribute__((ext_vector_type(8)))  float    v8f;

#define NPG   256        // nodes per graph
#define BGR   64         // graphs
#define KNN   100
#define F_IN  5
#define H     128
#define H2    768
#define LDH   136        // row-major stride: [node][feat], 272B rows (17x16B)
#define LDT   272        // feat-major stride: [feat][node], 544B rows (34x16B)
#define LDG   776        // MLP activation stride, 1552B rows (97x16B)
#define SLOPE 0.01f

#define WMMA_F16(a, b, c) \
  __builtin_amdgcn_wmma_f32_16x16x32_f16(false, (a), false, (b), (short)0, (c), false, false)

static __device__ __forceinline__ float leaky_f(float v) {
  return v > 0.f ? v : SLOPE * v;
}

// ---- WMMA fragment loaders (CDNA5 wave32 layouts, 16x16x32 f16) ------------

// B-matrix (32x16, KxN) where storage is [N][K]-contiguous with stride ld:
// per lane: row n = lane&15, K = (lane>>4)*8 + 0..7 and (lane>>4)*8 + 16..23.
static __device__ __forceinline__ v16h wmma_b_row(const half_t* base, int ld) {
  int lane = threadIdx.x & 31;
  int n  = lane & 15;
  int kb = (lane >> 4) * 8;
  const v8h* p = (const v8h*)(base + n * ld + kb);
  v8h lo = p[0];
  v8h hi = p[2];
  return __builtin_shufflevector(lo, hi, 0, 1, 2, 3, 4, 5, 6, 7,
                                         8, 9, 10, 11, 12, 13, 14, 15);
}

// Fragment from a pre-swizzled global f16 tile (512 f16 = 1KB, per-lane 32B).
static __device__ __forceinline__ v16h wmma_frag_sw(const half_t* tile) {
  int lane = threadIdx.x & 31;
  const v16h* p = (const v16h*)tile;
  return p[lane];
}

// A-matrix (16x32) from LDS row-major [M][K], stride ld (used by MLP).
static __device__ __forceinline__ v16h wmma_a_lds(const half_t* base, int ld) {
  int lane = threadIdx.x & 31;
  int m  = lane & 15;
  int kb = (lane >> 4) * 8;
  const v8h* p = (const v8h*)(base + m * ld + kb);
  v8h lo = p[0];   // K = kb + 0..7
  v8h hi = p[2];   // K = kb + 16..23
  return __builtin_shufflevector(lo, hi, 0, 1, 2, 3, 4, 5, 6, 7,
                                         8, 9, 10, 11, 12, 13, 14, 15);
}

// ---- adjacency expansion via 256-entry LDS LUT (4KB) ------------------------
// LUT[i] = 8 f16 with element j = (i>>j)&1 ? 1.0 : 0.0

static __device__ __forceinline__ void build_adj_lut(half_t* lut) {
  int t = threadIdx.x;               // 256 threads -> one 16B entry each
  v8h e;
#pragma unroll
  for (int j = 0; j < 8; ++j) e[j] = ((t >> j) & 1) ? (half_t)1.f : (half_t)0.f;
  *(v8h*)(lut + t * 8) = e;
}

// A-fragment from the adjacency bitmask: one dword, two byte-indexed LUT loads.
static __device__ __forceinline__ v16h wmma_a_adj(const unsigned* bits, const half_t* lut,
                                                  int row0, int k0) {
  int lane = threadIdx.x & 31;
  int m  = row0 + (lane & 15);
  int sh = (lane >> 4) * 8;
  unsigned w  = bits[m * 8 + (k0 >> 5)];
  unsigned b0 = (w >> sh) & 0xFFu;          // elements 0..7  (K = k0+sh+0..7)
  unsigned b1 = (w >> (sh + 16)) & 0xFFu;   // elements 8..15 (K = k0+sh+16..23)
  v8h lo = *(const v8h*)(lut + b0 * 8);
  v8h hi = *(const v8h*)(lut + b1 * 8);
  return __builtin_shufflevector(lo, hi, 0, 1, 2, 3, 4, 5, 6, 7,
                                         8, 9, 10, 11, 12, 13, 14, 15);
}

// ---- kNN: per-row 100-nearest via bit-greedy threshold on f32 bit patterns --

static __device__ __forceinline__ void knn_build(unsigned* Abits, const float* xs) {
  int wave = threadIdx.x >> 5;
  int lane = threadIdx.x & 31;
  for (int i = wave; i < NPG; i += 8) {
    float xi[F_IN];
#pragma unroll
    for (int c = 0; c < F_IN; ++c) xi[c] = xs[i * 8 + c];
    unsigned du[8];
#pragma unroll
    for (int r = 0; r < 8; ++r) {
      int j = r * 32 + lane;
      float d = 0.f;
#pragma unroll
      for (int c = 0; c < F_IN; ++c) {
        float t = xs[j * 8 + c] - xi[c];
        d += t * t;
      }
      unsigned u = __float_as_uint(d);
      if (j == i) u = 0x7F800000u;   // exclude self (inf)
      du[r] = u;
    }
    unsigned t = 0u;
    for (int bit = 30; bit >= 0; --bit) {
      unsigned cand = t | (1u << bit);
      int cnt = 0;
#pragma unroll
      for (int r = 0; r < 8; ++r)
        cnt += __builtin_popcount(__builtin_amdgcn_ballot_w32(du[r] < cand));
      if (cnt < KNN) t = cand;
    }
#pragma unroll
    for (int r = 0; r < 8; ++r) {
      unsigned word = __builtin_amdgcn_ballot_w32(du[r] <= t);
      if (lane == 0) Abits[i * 8 + r] = word;
    }
  }
}

// ---- aggregation: g = 0.01 * A_adj @ h ; input hT (feat-major) -------------
// Dual store: gT (vector 16B stores, may be null) and gRow (scalar transposes).

template <int CT>  // feature tiles: 2 (32 feats) or 8 (128 feats)
static __device__ __forceinline__ void aggregate_T(const unsigned* Abits, const half_t* lut,
                                                   const half_t* hT,
                                                   half_t* gRow, half_t* gT) {
  int wave = threadIdx.x >> 5;
  int lane = threadIdx.x & 31;
  for (int tr = wave; tr < 16; tr += 8) {          // node row tiles
    v8f z = {};
    v8f acc[CT];
#pragma unroll
    for (int tc = 0; tc < CT; ++tc) acc[tc] = z;
#pragma unroll
    for (int ks = 0; ks < 8; ++ks) {               // K = 256 nodes
      v16h a = wmma_a_adj(Abits, lut, tr * 16, ks * 32);
#pragma unroll
      for (int tc = 0; tc < CT; ++tc) {
        v16h b = wmma_b_row(hT + (tc * 16) * LDT + ks * 32, LDT);
        acc[tc] = WMMA_F16(a, b, acc[tc]);
      }
    }
    int mb = (lane >> 4) * 8;
    int nn = lane & 15;
#pragma unroll
    for (int tc = 0; tc < CT; ++tc) {
      v8h hv;
#pragma unroll
      for (int r = 0; r < 8; ++r) {
        float v = acc[tc][r] * 0.01f;
        hv[r] = (half_t)v;
        gRow[(tr * 16 + mb + r) * LDH + tc * 16 + nn] = (half_t)v;
      }
      if (gT) *(v8h*)(gT + (tc * 16 + nn) * LDT + tr * 16 + mb) = hv;
    }
  }
}

// ---- transform phase A: partial[node][outfeat] = h@W0 + g1@W1 ---------------
// Computed transposed: D = W_w^T @ h_w^T ; A = swizzled W^T fragments (global),
// B = row-major h/g1 (contiguous LDS loads). Vector stores.

template <int KS>
static __device__ __forceinline__ void xform_partial(const half_t* B0, const half_t* B1,
                                                     const half_t* WT, half_t* p) {
  int wave = threadIdx.x >> 5;
  int lane = threadIdx.x & 31;
  int mt = wave;                                   // out-feat tile 0..7
  int mb = (lane >> 4) * 8;
  int nn = lane & 15;
  const half_t* Bs[2] = {B0, B1};
#pragma unroll
  for (int ch = 0; ch < 2; ++ch) {                 // node tiles in 2 chunks of 8
    v8f z = {};
    v8f acc[8];
#pragma unroll
    for (int nc = 0; nc < 8; ++nc) acc[nc] = z;
#pragma unroll
    for (int w = 0; w < 2; ++w) {
      const half_t* WTt = WT + (size_t)((w * 8 + mt) * KS) * 512;
#pragma unroll
      for (int ks = 0; ks < KS; ++ks) {
        __builtin_prefetch(WTt + (size_t)(ks + 1) * 512, 0, 1);
        v16h a = wmma_frag_sw(WTt + (size_t)ks * 512);
#pragma unroll
        for (int nc = 0; nc < 8; ++nc) {
          v16h b = wmma_b_row(Bs[w] + (ch * 128 + nc * 16) * LDH + ks * 32, LDH);
          acc[nc] = WMMA_F16(a, b, acc[nc]);
        }
      }
    }
#pragma unroll
    for (int nc = 0; nc < 8; ++nc) {
      v8h hv;
#pragma unroll
      for (int r = 0; r < 8; ++r) hv[r] = (half_t)acc[nc][r];
      *(v8h*)(p + (ch * 128 + nc * 16 + nn) * LDH + mt * 16 + mb) = hv;
    }
  }
}

// ---- transform phase B: hnew = leaky(partial + g2@W2 + bias) ----------------
// Dual store: hRow (vector) + hT (scalar transposing stores).

template <int KS>
static __device__ __forceinline__ void xform_final(const half_t* B2, const half_t* WT,
                                                   const half_t* p, const float* bias,
                                                   half_t* hRow, half_t* hT) {
  int wave = threadIdx.x >> 5;
  int lane = threadIdx.x & 31;
  int mt = wave;
  int mb = (lane >> 4) * 8;
  int nn = lane & 15;
  float bcol[8];
#pragma unroll
  for (int r = 0; r < 8; ++r) bcol[r] = bias[mt * 16 + mb + r];
  const half_t* WTt = WT + (size_t)((2 * 8 + mt) * KS) * 512;   // w = 2 block
#pragma unroll
  for (int ch = 0; ch < 2; ++ch) {
    v8f acc[8];
#pragma unroll
    for (int nc = 0; nc < 8; ++nc) {
      v8h hv = *(const v8h*)(p + (ch * 128 + nc * 16 + nn) * LDH + mt * 16 + mb);
#pragma unroll
      for (int r = 0; r < 8; ++r) acc[nc][r] = (float)hv[r];
    }
#pragma unroll
    for (int ks = 0; ks < KS; ++ks) {
      __builtin_prefetch(WTt + (size_t)(ks + 1) * 512, 0, 1);
      v16h a = wmma_frag_sw(WTt + (size_t)ks * 512);
#pragma unroll
      for (int nc = 0; nc < 8; ++nc) {
        v16h b = wmma_b_row(B2 + (ch * 128 + nc * 16) * LDH + ks * 32, LDH);
        acc[nc] = WMMA_F16(a, b, acc[nc]);
      }
    }
#pragma unroll
    for (int nc = 0; nc < 8; ++nc) {
      v8h hv;
#pragma unroll
      for (int r = 0; r < 8; ++r) {
        float v = leaky_f(acc[nc][r] + bcol[r]);
        hv[r] = (half_t)v;
        hT[(mt * 16 + mb + r) * LDT + ch * 128 + nc * 16 + nn] = (half_t)v;
      }
      *(v8h*)(hRow + (ch * 128 + nc * 16 + nn) * LDH + mt * 16 + mb) = hv;
    }
  }
}

// ---- per-graph mean+max pool over 256 nodes, reading feat-major hT ---------

static __device__ __forceinline__ void pool_write_T(const half_t* hT, float* scratch,
                                                    float* grow, int off) {
  int tid = threadIdx.x;
  int f  = tid & 127;
  int hi = tid >> 7;
  const v8h* row = (const v8h*)(hT + f * LDT + hi * 128);
  float s = 0.f, m = -3.4e38f;
#pragma unroll
  for (int c = 0; c < 16; ++c) {
    v8h v = row[c];
#pragma unroll
    for (int r = 0; r < 8; ++r) {
      float x = (float)v[r];
      s += x;
      m = fmaxf(m, x);
    }
  }
  if (hi) { scratch[f] = s; scratch[128 + f] = m; }
  __syncthreads();
  if (!hi) {
    s += scratch[f];
    m = fmaxf(m, scratch[128 + f]);
    grow[off + f]     = s * (1.f / NPG);
    grow[off + H + f] = m;
  }
  __syncthreads();
}

// ============================ Kernel 0a: W^T swizzle (A-fragment order) ======

__global__ __launch_bounds__(256)
void swizzle_wT(const float* __restrict__ src, half_t* __restrict__ dst,
                int nw, int nmt, int nks, int krows, int ncols) {
  int total = nw * nmt * nks * 512;
  for (int i = blockIdx.x * blockDim.x + threadIdx.x; i < total;
       i += gridDim.x * blockDim.x) {
    int e    = i & 15;
    int lane = (i >> 4) & 31;
    int t    = i >> 9;
    int ks = t % nks;
    int mt = (t / nks) % nmt;
    int w  = t / (nks * nmt);
    int k = ks * 32 + (lane >> 4) * 8 + ((e < 8) ? e : (e + 8));
    int m = mt * 16 + (lane & 15);
    float v = (k < krows) ? src[((size_t)w * krows + k) * ncols + m] : 0.f;
    dst[i] = (half_t)v;
  }
}

// ============================ Kernel 0b: W swizzle (B-fragment order) ========

__global__ __launch_bounds__(256)
void swizzle_w(const float* __restrict__ src, half_t* __restrict__ dst,
               int nw, int nks, int ntc, int krows, int ncols) {
  int total = nw * nks * ntc * 512;
  for (int i = blockIdx.x * blockDim.x + threadIdx.x; i < total;
       i += gridDim.x * blockDim.x) {
    int e    = i & 15;
    int lane = (i >> 4) & 31;
    int t    = i >> 9;
    int tc = t % ntc;
    int ks = (t / ntc) % nks;
    int w  = t / (ntc * nks);
    int k = ks * 32 + (lane >> 4) * 8 + ((e < 8) ? e : (e + 8));
    int n = tc * 16 + (lane & 15);
    float v = (k < krows) ? src[((size_t)w * krows + k) * ncols + n] : 0.f;
    dst[i] = (half_t)v;
  }
}

// ============================ Kernel 1: per-graph GNN ========================

__global__ __launch_bounds__(256)
void gnn_graph_kernel(const float* __restrict__ x,
                      const half_t* __restrict__ wT1, const float* __restrict__ c1b,
                      const half_t* __restrict__ wT2, const float* __restrict__ c2b,
                      const half_t* __restrict__ wT3, const float* __restrict__ c3b,
                      float* __restrict__ g) {
  extern __shared__ char smem[];
  unsigned* Abits = (unsigned*)smem;                  // 8 KB bitmask adjacency
  half_t*   lut   = (half_t*)(smem + 8192);           // 4 KB byte->8xf16 LUT
  float*    pscr  = (float*)(smem + 12288);           // 1 KB pool scratch
  half_t*   U1    = (half_t*)(smem + 13312);          // 4 x 69632 B units
  half_t*   U2    = U1 + NPG * LDH;
  half_t*   U3    = U2 + NPG * LDH;
  half_t*   U4    = U3 + NPG * LDH;

  half_t* PH = U1;   // h row-major
  half_t* PT = U2;   // h feat-major
  half_t* PG = U3;   // g1 row-major (later: new hT)
  half_t* PQ = U4;   // g1 feat-major (later: partial)
  float*  xs = (float*)U4;   // kNN staging overlaps PQ (dead until aggregate #1)

  int bg = blockIdx.x;
  const float* xg = x + (size_t)bg * NPG * F_IN;
  float* grow = g + (size_t)bg * H2;

  build_adj_lut(lut);

  // stage x: xs (f32 padded-8), PH row-major padded-32, PT feat-major
  for (int idx = threadIdx.x; idx < NPG * 8; idx += blockDim.x) {
    int n = idx >> 3, c = idx & 7;
    xs[idx] = (c < F_IN) ? xg[n * F_IN + c] : 0.f;
  }
  for (int idx = threadIdx.x; idx < NPG * 32; idx += blockDim.x) {
    int n = idx >> 5, c = idx & 31;
    PH[n * LDH + c] = (half_t)((c < F_IN) ? xg[n * F_IN + c] : 0.f);
  }
  for (int idx = threadIdx.x; idx < 32 * NPG; idx += blockDim.x) {
    int f = idx >> 8, n = idx & 255;
    PT[f * LDT + n] = (half_t)((f < F_IN) ? xg[n * F_IN + f] : 0.f);
  }
  __syncthreads();

  knn_build(Abits, xs);
  __syncthreads();

  // -------- layer 1 (feat width padded to 32, weights pre-padded) ----------
  aggregate_T<2>(Abits, lut, PT, PG, PQ);            __syncthreads();
  aggregate_T<2>(Abits, lut, PQ, PT, nullptr);       __syncthreads();  // g2 -> PT
  xform_partial<1>(PH, PG, wT1, PQ);                 __syncthreads();
  xform_final<1>(PT, wT1, PQ, c1b, PH, PG);          __syncthreads();
  pool_write_T(PG, pscr, grow, 0);
  { half_t* t = PT; PT = PG; PG = t; }               // h: PH row + PT feat-major

  // -------- layer 2 --------------------------------------------------------
  aggregate_T<8>(Abits, lut, PT, PG, PQ);            __syncthreads();
  aggregate_T<8>(Abits, lut, PQ, PT, nullptr);       __syncthreads();
  xform_partial<4>(PH, PG, wT2, PQ);                 __syncthreads();
  xform_final<4>(PT, wT2, PQ, c2b, PH, PG);          __syncthreads();
  pool_write_T(PG, pscr, grow, 2 * H);
  { half_t* t = PT; PT = PG; PG = t; }

  // -------- layer 3 --------------------------------------------------------
  aggregate_T<8>(Abits, lut, PT, PG, PQ);            __syncthreads();
  aggregate_T<8>(Abits, lut, PQ, PT, nullptr);       __syncthreads();
  xform_partial<4>(PH, PG, wT3, PQ);                 __syncthreads();
  xform_final<4>(PT, wT3, PQ, c3b, PH, PG);          __syncthreads();
  pool_write_T(PG, pscr, grow, 4 * H);
}

// ============================ Kernel 2: BN + MLP head ========================

static __device__ __forceinline__ void mlp_layer(const half_t* pa, half_t* pb,
                                                 const half_t* Wsw, const float* bias) {
  int wave = threadIdx.x >> 5;
  int lane = threadIdx.x & 31;
  int tr   = wave & 3;                 // 4 row tiles of the 64-row activation
  int tch0 = (wave >> 2) * 24;         // waves 0-3: cols 0-23, waves 4-7: 24-47
#pragma unroll
  for (int chunk = 0; chunk < 3; ++chunk) {
    int tcb = tch0 + chunk * 8;
    v8f z = {};
    v8f acc[8];
#pragma unroll
    for (int tc = 0; tc < 8; ++tc) acc[tc] = z;
    for (int ks = 0; ks < 24; ++ks) {
      __builtin_prefetch(Wsw + ((size_t)((ks + 1) * 48) + tcb) * 512, 0, 1);
      v16h a = wmma_a_lds(pa + tr * 16 * LDG + ks * 32, LDG);
#pragma unroll
      for (int tc = 0; tc < 8; ++tc) {
        v16h b = wmma_frag_sw(Wsw + ((size_t)(ks * 48) + tcb + tc) * 512);
        acc[tc] = WMMA_F16(a, b, acc[tc]);
      }
    }
    int mb = (lane >> 4) * 8;
    int nn = lane & 15;
#pragma unroll
    for (int tc = 0; tc < 8; ++tc) {
      int n = (tcb + tc) * 16 + nn;
      float bv = bias[n];
#pragma unroll
      for (int r = 0; r < 8; ++r)
        pb[(tr * 16 + mb + r) * LDG + n] = (half_t)leaky_f(acc[tc][r] + bv);
    }
  }
}

__global__ __launch_bounds__(256)
void mlp_head_kernel(const float* __restrict__ g,
                     const float* __restrict__ gamma, const float* __restrict__ beta,
                     const half_t* __restrict__ linws, const float* __restrict__ linb,
                     const float* __restrict__ outw,   const float* __restrict__ outb,
                     float* __restrict__ out) {
  extern __shared__ char smem[];
  half_t* gA = (half_t*)smem;                  // 64 x 776 f16
  half_t* gB = gA + BGR * LDG;

  for (int col = threadIdx.x; col < H2; col += blockDim.x) {
    float s = 0.f, sq = 0.f;
    for (int r = 0; r < BGR; ++r) {
      float v = g[(size_t)r * H2 + col];
      s += v; sq += v * v;
    }
    float mu  = s * (1.f / BGR);
    float var = sq * (1.f / BGR) - mu * mu;
    float sc  = rsqrtf(var + 1e-5f) * gamma[col];
    float sh  = beta[col] - mu * sc;
    for (int r = 0; r < BGR; ++r)
      gA[r * LDG + col] = (half_t)(g[(size_t)r * H2 + col] * sc + sh);
  }
  __syncthreads();

  half_t* pa = gA;
  half_t* pb = gB;
  for (int i = 0; i < 5; ++i) {
    mlp_layer(pa, pb, linws + (size_t)i * (24 * 48 * 512), linb + (size_t)i * H2);
    __syncthreads();
    half_t* tmp = pa; pa = pb; pb = tmp;
  }

  if (threadIdx.x < BGR) {
    int r = threadIdx.x;
    float s = outb[0];
    for (int k = 0; k < H2; ++k) s += (float)pa[r * LDG + k] * outw[k];
    out[r] = s;
  }
}

// ============================ launch =========================================

extern "C" void kernel_launch(void* const* d_in, const int* in_sizes, int n_in,
                              void* d_out, int out_size, void* d_ws, size_t ws_size,
                              hipStream_t stream) {
  (void)in_sizes; (void)n_in; (void)out_size; (void)ws_size;
  const float* x    = (const float*)d_in[0];
  const float* c1w  = (const float*)d_in[2];
  const float* c1b  = (const float*)d_in[3];
  const float* c2w  = (const float*)d_in[4];
  const float* c2b  = (const float*)d_in[5];
  const float* c3w  = (const float*)d_in[6];
  const float* c3b  = (const float*)d_in[7];
  const float* gam  = (const float*)d_in[8];
  const float* bet  = (const float*)d_in[9];
  const float* linw = (const float*)d_in[10];
  const float* linb = (const float*)d_in[11];
  const float* outw = (const float*)d_in[12];
  const float* outb = (const float*)d_in[13];
  float* out = (float*)d_out;

  // workspace: [ g: 64x768 f32 | swizzled f16 weights ]
  float*  g   = (float*)d_ws;
  half_t* wsH = (half_t*)((char*)d_ws + (size_t)BGR * H2 * sizeof(float));
  half_t* wT1 = wsH;                       // 3*8*1 tiles  = 12288 f16 (A-order, K padded to 32)
  half_t* wT2 = wT1 + 3 * 8 * 1 * 512;     // 3*8*4 tiles  = 49152 f16
  half_t* wT3 = wT2 + 3 * 8 * 4 * 512;     // 3*8*4 tiles  = 49152 f16
  half_t* wLs = wT3 + 3 * 8 * 4 * 512;     // 5*24*48 tiles = 2949120 f16 (B-order)

  swizzle_wT<<<(3 * 8 * 1 * 512 + 255) / 256, 256, 0, stream>>>(c1w, wT1, 3, 8, 1, F_IN, H);
  swizzle_wT<<<(3 * 8 * 4 * 512 + 255) / 256, 256, 0, stream>>>(c2w, wT2, 3, 8, 4, H, H);
  swizzle_wT<<<(3 * 8 * 4 * 512 + 255) / 256, 256, 0, stream>>>(c3w, wT3, 3, 8, 4, H, H);
  swizzle_w <<<(5 * 24 * 48 * 512 + 255) / 256, 256, 0, stream>>>(linw, wLs, 5, 24, 48, H2, H2);

  size_t smem1 = 13312 + (size_t)4 * NPG * LDH * sizeof(half_t);  // 291840 B
  size_t smem2 = (size_t)2 * BGR * LDG * sizeof(half_t);          // 198656 B

  gnn_graph_kernel<<<BGR, 256, smem1, stream>>>(x, wT1, c1b, wT2, c2b, wT3, c3b, g);
  mlp_head_kernel<<<1, 256, smem2, stream>>>(g, gam, bet, wLs, linb, outw, outb, out);
}
